// TreeGNNNodeSelectionPolicy_57827439674229
// MI455X (gfx1250) — matrix operand order repaired and checked by reading, not compiled, not run on hardware
//
#include <hip/hip_runtime.h>
#include <hip/hip_fp16.h>

typedef __attribute__((ext_vector_type(16))) _Float16 v16h;
typedef __attribute__((ext_vector_type(8)))  _Float16 v8h;
typedef __attribute__((ext_vector_type(8)))  float    v8f;

#define HD    64
#define HP    (HD + 8)   // padded f16 row stride: 144 B, 16B-aligned, bank-rotating
#define FIN   19
#define NPB   128        // nodes per block (8 waves x 16 rows)
#define LNEPS 1e-5f

__device__ __forceinline__ float leaky(float v) { return v > 0.f ? v : 0.01f * v; }

__device__ __forceinline__ v16h cat16(v8h lo, v8h hi) {
    return __builtin_shufflevector(lo, hi, 0,1,2,3,4,5,6,7,8,9,10,11,12,13,14,15);
}
// A fragment (16x32 f16): per-lane one contiguous 32B run of K at k-offset ak
__device__ __forceinline__ v16h afrag(const _Float16* row, int ak) {
    v8h lo = *(const v8h*)(row + ak);
    v8h hi = *(const v8h*)(row + ak + 8);
    return cat16(lo, hi);
}
// B fragment (32x16 f16): per-lane one contiguous 32B run of K for a fixed column
__device__ __forceinline__ v16h bfrag(const _Float16* row, int bk) {
    v8h lo = *(const v8h*)(row + bk);
    v8h hi = *(const v8h*)(row + bk + 8);
    return cat16(lo, hi);
}

// ---------------------------------------------------------------------------
// Kernel 1: h = leaky(x@pW+pb) (WMMA, K zero-padded 19->32), then
// 2x { LayerNorm(h + leaky(h@eW+eb)) } with 64x64 WMMA GEMMs. h -> global f16.
// ---------------------------------------------------------------------------
__global__ __launch_bounds__(256)
void tgnn_embed_kernel(const float* __restrict__ x,
                       const float* __restrict__ pW, const float* __restrict__ pb,
                       const float* __restrict__ eW, const float* __restrict__ eb,
                       const float* __restrict__ gma, const float* __restrict__ bta,
                       _Float16* __restrict__ gh, int n)
{
    __shared__ alignas(16) _Float16 s_h[NPB][HP];    // node tile (f16)
    __shared__ alignas(16) _Float16 s_d[NPB][HP];    // branch output (f16)
    __shared__ alignas(16) _Float16 s_w[HD][HP];     // transposed weight WT[n][k]
    __shared__ alignas(16) _Float16 s_x[NPB][40];    // x tile, K padded to 32
    __shared__ float s_b[HD], s_g[HD], s_be[HD];

    const int  tid   = threadIdx.x;
    const int  lane  = tid & 31;
    const int  wave  = tid >> 5;
    const int  nodeL = tid >> 1;             // 0..127  (wave == nodeL>>4)
    const int  half  = tid & 1;
    const long long node   = (long long)blockIdx.x * NPB + nodeL;
    const long long nclamp = (node < n) ? node : 0;
    const int  r0 = wave * 16;

    // stage pW^T (64 x 32, zero-padded K) and x tile (f16, zero-padded K)
    for (int i = tid; i < HD * 32; i += 256) {
        int k = i & 31, nn = i >> 5;
        s_w[nn][k] = (k < FIN) ? (_Float16)pW[k * HD + nn] : (_Float16)0.f;
    }
    if (tid < HD) s_b[tid] = pb[tid];
    #pragma unroll
    for (int kk = 0; kk < 16; ++kk) {
        int k = half * 16 + kk;
        s_x[nodeL][k] = (k < FIN) ? (_Float16)x[nclamp * FIN + k] : (_Float16)0.f;
    }
    __syncthreads();

    // ---- input projection via WMMA (single K-step) ----
    {
        v8f c[4] = {};
        const int arow = r0 + (lane & 15);
        const int ak   = (lane < 16) ? 0 : 16;
        v16h a = afrag(&s_x[arow][0], ak);
        #pragma unroll
        for (int nt = 0; nt < 4; ++nt) {
            const int bn = nt * 16 + (lane & 15);
            const int bk = (lane < 16) ? 0 : 16;
            v16h b = bfrag(&s_w[bn][0], bk);
            c[nt] = __builtin_amdgcn_wmma_f32_16x16x32_f16(
                        false, a, false, b, (short)0, c[nt], false, false);
        }
        #pragma unroll
        for (int nt = 0; nt < 4; ++nt)
            #pragma unroll
            for (int r = 0; r < 8; ++r) {
                int row = r0 + r + ((lane < 16) ? 0 : 8);
                int col = nt * 16 + (lane & 15);
                s_h[row][col] = (_Float16)leaky(c[nt][r] + s_b[col]);
            }
    }

    // ---- two residual + LayerNorm layers ----
    for (int layer = 0; layer < 2; ++layer) {
        __syncthreads();                      // all waves done with s_w
        const float* W = eW + layer * HD * HD;
        for (int i = tid; i < HD * HD; i += 256) {
            int k = i >> 6, nn = i & 63;
            s_w[nn][k] = (_Float16)W[i];      // transpose: WT[n][k]
        }
        if (tid < HD) {
            s_b[tid]  = eb[layer * HD + tid];
            s_g[tid]  = gma[layer * HD + tid];
            s_be[tid] = bta[layer * HD + tid];
        }
        __syncthreads();

        // WMMA: wave-private 16-row tile, 2 K-steps x 4 N-tiles
        v8f c[4] = {};
        #pragma unroll
        for (int ks = 0; ks < 2; ++ks) {
            const int k0   = ks * 32;
            const int arow = r0 + (lane & 15);
            const int ak   = k0 + ((lane < 16) ? 0 : 16);
            v16h a = afrag(&s_h[arow][0], ak);
            #pragma unroll
            for (int nt = 0; nt < 4; ++nt) {
                const int bn = nt * 16 + (lane & 15);
                const int bk = k0 + ((lane < 16) ? 0 : 16);
                v16h b = bfrag(&s_w[bn][0], bk);
                c[nt] = __builtin_amdgcn_wmma_f32_16x16x32_f16(
                            false, a, false, b, (short)0, c[nt], false, false);
            }
        }
        // epilogue: store branch output only (residual folded in LN pass)
        #pragma unroll
        for (int nt = 0; nt < 4; ++nt)
            #pragma unroll
            for (int r = 0; r < 8; ++r) {
                int row = r0 + r + ((lane < 16) ? 0 : 8);
                int col = nt * 16 + (lane & 15);
                s_d[row][col] = (_Float16)leaky(c[nt][r] + s_b[col]);
            }
        __syncthreads();

        // LayerNorm( h + d ): 2 threads/node, vectorized b128 LDS traffic
        {
            float v[32], sum = 0.f, sq = 0.f;
            #pragma unroll
            for (int q = 0; q < 4; ++q) {
                v8h hh = *(const v8h*)&s_h[nodeL][half * 32 + q * 8];
                v8h dd = *(const v8h*)&s_d[nodeL][half * 32 + q * 8];
                #pragma unroll
                for (int i = 0; i < 8; ++i) {
                    float t = (float)hh[i] + (float)dd[i];
                    v[q * 8 + i] = t; sum += t; sq += t * t;
                }
            }
            sum += __shfl_xor(sum, 1);
            sq  += __shfl_xor(sq, 1);
            float mu  = sum * (1.f / 64.f);
            float var = sq * (1.f / 64.f) - mu * mu;
            float rs  = rsqrtf(var + LNEPS);
            #pragma unroll
            for (int q = 0; q < 4; ++q) {
                v8h o;
                #pragma unroll
                for (int i = 0; i < 8; ++i) {
                    int col = half * 32 + q * 8 + i;
                    o[i] = (_Float16)((v[q * 8 + i] - mu) * rs * s_g[col] + s_be[col]);
                }
                *(v8h*)&s_h[nodeL][half * 32 + q * 8] = o;
            }
        }
    }

    if (node < n) {
        #pragma unroll
        for (int q = 0; q < 4; ++q)
            *(v8h*)(gh + node * HD + half * 32 + q * 8) =
                *(const v8h*)&s_h[nodeL][half * 32 + q * 8];
    }
}

// ---------------------------------------------------------------------------
// Kernel 2: edge scatter: one wave per edge, 2 features per lane,
// f32 atomics into sum[parent]; h gathers are L2-resident (f16 state fits L2).
// ---------------------------------------------------------------------------
__global__ __launch_bounds__(256)
void tgnn_scatter_kernel(const long long* __restrict__ ei,
                         const _Float16* __restrict__ gh,
                         float* __restrict__ gsum, float* __restrict__ gcnt,
                         int e)
{
    const int lane = threadIdx.x & 31;
    const long long nw = (long long)gridDim.x * 8;
    for (long long edge = (long long)blockIdx.x * 8 + (threadIdx.x >> 5);
         edge < e; edge += nw) {
        if (edge + nw < e) {
            __builtin_prefetch(&ei[edge + nw], 0, 1);       // global_prefetch_b8
            __builtin_prefetch(&ei[e + edge + nw], 0, 1);
        }
        long long parent = ei[edge];
        long long child  = ei[e + edge];
        __half2 hv = ((const __half2*)gh)[child * 32 + lane];
        float2  f  = __half22float2(hv);
        atomicAdd(&gsum[parent * HD + lane * 2], f.x);
        atomicAdd(&gsum[parent * HD + lane * 2 + 1], f.y);
        if (lane == 0) atomicAdd(&gcnt[parent], 1.f);
    }
}

// ---------------------------------------------------------------------------
// Kernel 3: mean = sum/max(cnt,1); delta = leaky(mean@W1+b1)@W2+b2;
// h += mask ? delta : 0.  Consumes and re-zeroes sum/cnt for the next round.
// ---------------------------------------------------------------------------
__global__ __launch_bounds__(256)
void tgnn_update_kernel(const float* __restrict__ W1, const float* __restrict__ b1,
                        const float* __restrict__ W2, const float* __restrict__ b2,
                        float* __restrict__ gsum, float* __restrict__ gcnt,
                        _Float16* __restrict__ gh, int n)
{
    __shared__ alignas(16) _Float16 s_m[NPB][HP];    // mean -> reused for delta
    __shared__ alignas(16) _Float16 s_z[NPB][HP];    // stage-1 output
    __shared__ alignas(16) _Float16 s_w1[HD][HP], s_w2[HD][HP];
    __shared__ float s_b1[HD], s_b2[HD], s_cnt[NPB];

    const int  tid   = threadIdx.x;
    const int  lane  = tid & 31;
    const int  wave  = tid >> 5;
    const int  nodeL = tid >> 1;
    const int  half  = tid & 1;
    const long long node   = (long long)blockIdx.x * NPB + nodeL;
    const long long nclamp = (node < n) ? node : 0;
    const int  r0 = wave * 16;

    for (int i = tid; i < HD * HD; i += 256) {
        int k = i >> 6, nn = i & 63;
        s_w1[nn][k] = (_Float16)W1[i];
        s_w2[nn][k] = (_Float16)W2[i];
    }
    if (tid < HD) { s_b1[tid] = b1[tid]; s_b2[tid] = b2[tid]; }
    __syncthreads();

    // ---- mean (vectorized) + consume/zero accumulators ----
    {
        float cv  = gcnt[nclamp];
        float inv = 1.f / fmaxf(cv, 1.f);
        const float4* sp = (const float4*)(gsum + nclamp * HD + half * 32);
        #pragma unroll
        for (int q = 0; q < 4; ++q) {
            float4 a = sp[q * 2], bq = sp[q * 2 + 1];
            v8h o;
            o[0] = (_Float16)(a.x * inv);  o[1] = (_Float16)(a.y * inv);
            o[2] = (_Float16)(a.z * inv);  o[3] = (_Float16)(a.w * inv);
            o[4] = (_Float16)(bq.x * inv); o[5] = (_Float16)(bq.y * inv);
            o[6] = (_Float16)(bq.z * inv); o[7] = (_Float16)(bq.w * inv);
            *(v8h*)&s_m[nodeL][half * 32 + q * 8] = o;
        }
        if (half == 0) s_cnt[nodeL] = cv;
        if (node < n) {
            float4* zp = (float4*)(gsum + node * HD + half * 32);
            #pragma unroll
            for (int q = 0; q < 8; ++q) zp[q] = make_float4(0.f, 0.f, 0.f, 0.f);
            if (half == 0) gcnt[node] = 0.f;
        }
    }
    __syncthreads();

    // ---- stage 1: z = leaky(mean @ W1 + b1) ----
    {
        v8f c[4] = {};
        #pragma unroll
        for (int ks = 0; ks < 2; ++ks) {
            const int k0   = ks * 32;
            const int arow = r0 + (lane & 15);
            const int ak   = k0 + ((lane < 16) ? 0 : 16);
            v16h a = afrag(&s_m[arow][0], ak);
            #pragma unroll
            for (int nt = 0; nt < 4; ++nt) {
                const int bn = nt * 16 + (lane & 15);
                const int bk = k0 + ((lane < 16) ? 0 : 16);
                v16h b = bfrag(&s_w1[bn][0], bk);
                c[nt] = __builtin_amdgcn_wmma_f32_16x16x32_f16(
                            false, a, false, b, (short)0, c[nt], false, false);
            }
        }
        #pragma unroll
        for (int nt = 0; nt < 4; ++nt)
            #pragma unroll
            for (int r = 0; r < 8; ++r) {
                int row = r0 + r + ((lane < 16) ? 0 : 8);
                int col = nt * 16 + (lane & 15);
                s_z[row][col] = (_Float16)leaky(c[nt][r] + s_b1[col]);
            }
    }
    __syncthreads();

    // ---- stage 2: delta = z @ W2 + b2, masked; stored into s_m ----
    {
        v8f c[4] = {};
        #pragma unroll
        for (int ks = 0; ks < 2; ++ks) {
            const int k0   = ks * 32;
            const int arow = r0 + (lane & 15);
            const int ak   = k0 + ((lane < 16) ? 0 : 16);
            v16h a = afrag(&s_z[arow][0], ak);
            #pragma unroll
            for (int nt = 0; nt < 4; ++nt) {
                const int bn = nt * 16 + (lane & 15);
                const int bk = k0 + ((lane < 16) ? 0 : 16);
                v16h b = bfrag(&s_w2[bn][0], bk);
                c[nt] = __builtin_amdgcn_wmma_f32_16x16x32_f16(
                            false, a, false, b, (short)0, c[nt], false, false);
            }
        }
        #pragma unroll
        for (int nt = 0; nt < 4; ++nt)
            #pragma unroll
            for (int r = 0; r < 8; ++r) {
                int row = r0 + r + ((lane < 16) ? 0 : 8);
                int col = nt * 16 + (lane & 15);
                float d = c[nt][r] + s_b2[col];
                s_m[row][col] = (_Float16)((s_cnt[row] > 0.f) ? d : 0.f);
            }
    }
    __syncthreads();

    // ---- h += delta (vectorized global RMW) ----
    if (node < n) {
        #pragma unroll
        for (int q = 0; q < 4; ++q) {
            v8h hv = *(v8h*)(gh + node * HD + half * 32 + q * 8);
            v8h dm = *(const v8h*)&s_m[nodeL][half * 32 + q * 8];
            v8h o;
            #pragma unroll
            for (int i = 0; i < 8; ++i)
                o[i] = (_Float16)((float)hv[i] + (float)dm[i]);
            *(v8h*)(gh + node * HD + half * 32 + q * 8) = o;
        }
    }
}

// ---------------------------------------------------------------------------
// Kernel 4: score head on K candidates + softmax (single block, 1 thread/cand)
// ---------------------------------------------------------------------------
__global__ __launch_bounds__(512)
void tgnn_score_kernel(const long long* __restrict__ cand,
                       const _Float16* __restrict__ gh,
                       const float* __restrict__ hW1, const float* __restrict__ hb1,
                       const float* __restrict__ hW2, const float* __restrict__ hb2,
                       float* __restrict__ out)
{
    __shared__ float s_w[HD * HD];
    __shared__ float s_red[40];
    const int tid = threadIdx.x;
    const int nw  = blockDim.x >> 5;

    for (int i = tid; i < HD * HD; i += blockDim.x) s_w[i] = hW1[i];
    __syncthreads();

    long long node = cand[tid];
    float hv[HD];
    const v8h* hp = (const v8h*)(gh + node * HD);
    #pragma unroll
    for (int q = 0; q < 8; ++q) {
        v8h ch = hp[q];
        #pragma unroll
        for (int i = 0; i < 8; ++i) hv[q * 8 + i] = (float)ch[i];
    }

    float s = hb2[0];
    for (int j = 0; j < HD; ++j) {
        float acc = hb1[j];
        #pragma unroll
        for (int k = 0; k < HD; ++k) acc += hv[k] * s_w[k * HD + j];
        s += leaky(acc) * hW2[j];
    }

    // block max
    float m = s;
    for (int off = 16; off; off >>= 1) m = fmaxf(m, __shfl_xor(m, off));
    if ((tid & 31) == 0) s_red[tid >> 5] = m;
    __syncthreads();
    if (tid == 0) {
        float t = s_red[0];
        for (int i = 1; i < nw; ++i) t = fmaxf(t, s_red[i]);
        s_red[36] = t;
    }
    __syncthreads();
    float ex = __expf(s - s_red[36]);

    // block sum
    float ss = ex;
    for (int off = 16; off; off >>= 1) ss += __shfl_xor(ss, off);
    if ((tid & 31) == 0) s_red[tid >> 5] = ss;
    __syncthreads();
    if (tid == 0) {
        float t = 0.f;
        for (int i = 0; i < nw; ++i) t += s_red[i];
        s_red[37] = t;
    }
    __syncthreads();
    out[tid] = ex / s_red[37];
}

// ---------------------------------------------------------------------------
extern "C" void kernel_launch(void* const* d_in, const int* in_sizes, int n_in,
                              void* d_out, int out_size, void* d_ws, size_t ws_size,
                              hipStream_t stream)
{
    const float*     x    = (const float*)d_in[0];
    const long long* ei   = (const long long*)d_in[1];
    const long long* cand = (const long long*)d_in[2];
    const float* pW  = (const float*)d_in[3];
    const float* pb  = (const float*)d_in[4];
    const float* eW  = (const float*)d_in[5];
    const float* eb  = (const float*)d_in[6];
    const float* gma = (const float*)d_in[7];
    const float* bta = (const float*)d_in[8];
    const float* sW1 = (const float*)d_in[9];
    const float* sb1 = (const float*)d_in[10];
    const float* sW2 = (const float*)d_in[11];
    const float* sb2 = (const float*)d_in[12];
    const float* hW1 = (const float*)d_in[13];
    const float* hb1 = (const float*)d_in[14];
    const float* hW2 = (const float*)d_in[15];
    const float* hb2 = (const float*)d_in[16];

    const int n = in_sizes[0] / FIN;
    const int e = in_sizes[1] / 2;
    const int k = in_sizes[2];

    // workspace: h (f16) | sum (f32) | cnt (f32)
    char* ws = (char*)d_ws;
    size_t off = 0;
    _Float16* gh = (_Float16*)(ws + off);
    off += (size_t)n * HD * sizeof(_Float16);  off = (off + 255) & ~(size_t)255;
    float* gsum = (float*)(ws + off);
    off += (size_t)n * HD * sizeof(float);     off = (off + 255) & ~(size_t)255;
    float* gcnt = (float*)(ws + off);

    const int nblocks = (n + NPB - 1) / NPB;

    tgnn_embed_kernel<<<nblocks, 256, 0, stream>>>(x, pW, pb, eW, eb, gma, bta, gh, n);

    hipMemsetAsync(gsum, 0, (size_t)n * HD * sizeof(float), stream);
    hipMemsetAsync(gcnt, 0, (size_t)n * sizeof(float), stream);

    for (int it = 0; it < 3; ++it) {
        tgnn_scatter_kernel<<<4096, 256, 0, stream>>>(ei, gh, gsum, gcnt, e);
        tgnn_update_kernel<<<nblocks, 256, 0, stream>>>(
            sW1 + it * HD * HD, sb1 + it * HD,
            sW2 + it * HD * HD, sb2 + it * HD,
            gsum, gcnt, gh, n);
    }

    tgnn_score_kernel<<<1, k, 0, stream>>>(cand, gh, hW1, hb1, hW2, hb2, (float*)d_out);
}
